// SelfAttention_49572512531031
// MI455X (gfx1250) — matrix operand order repaired and checked by reading, not compiled, not run on hardware
//
#include <hip/hip_runtime.h>
#include <hip/hip_bf16.h>
#include <math.h>
#include <stdint.h>

#define N_TOK 8192
#define DIM   1024

typedef __bf16 bf16_t;
typedef __bf16 v16bf __attribute__((ext_vector_type(16)));
typedef __bf16 v8bf  __attribute__((ext_vector_type(8)));
typedef float  v8f   __attribute__((ext_vector_type(8)));
typedef int    v4i   __attribute__((ext_vector_type(4)));

__device__ __forceinline__ bf16_t f2bf(float f) { return (bf16_t)f; }
__device__ __forceinline__ float  bf2f(bf16_t b){ return (float)b; }

// ---- CDNA5 async global->LDS copy (ASYNCcnt DMA), guarded by __has_builtin ----
#if __has_builtin(__builtin_amdgcn_global_load_async_to_lds_b128) && \
    __has_builtin(__builtin_amdgcn_s_wait_asynccnt)
#define HAVE_ASYNC_LDS 1
#else
#define HAVE_ASYNC_LDS 0
#endif

#if HAVE_ASYNC_LDS
typedef __attribute__((address_space(1))) v4i glb_v4i;   // global b128 chunk
typedef __attribute__((address_space(3))) v4i lds_v4i;   // LDS b128 chunk
__device__ __forceinline__ glb_v4i* as_glb(const void* p) {
    return (glb_v4i*)(uintptr_t)p;
}
__device__ __forceinline__ lds_v4i* as_lds(const void* p) {
    // generic LDS pointer: low 32 bits are the LDS offset
    return (lds_v4i*)(uintptr_t)p;
}
#endif

// ---------------------------------------------------------------------------
// Small helper kernels
// ---------------------------------------------------------------------------

__global__ __launch_bounds__(256)
void colmean_f32(const float* __restrict__ x, float* __restrict__ mean,
                 int rows, int cols) {
    int c = blockIdx.x * blockDim.x + threadIdx.x;
    if (c >= cols) return;
    float s = 0.f;
    for (int r = 0; r < rows; ++r) s += x[(size_t)r * cols + c];
    mean[c] = s / (float)rows;
}

__global__ __launch_bounds__(256)
void prep_x(const float* __restrict__ x, const float* __restrict__ mean,
            bf16_t* __restrict__ xb, bf16_t* __restrict__ xcb,
            int rows, int cols) {
    size_t n = (size_t)rows * cols;
    size_t i = (size_t)blockIdx.x * blockDim.x + threadIdx.x;
    size_t stride = (size_t)gridDim.x * blockDim.x;
    for (; i < n; i += stride) {
        int c = (int)(i % cols);
        float v = x[i];
        xb[i]  = f2bf(v);
        xcb[i] = f2bf(v - mean[c]);
    }
}

// dst[c][r] = bf16(src[r][c])  (32x32 LDS-tiled transpose)
__global__ __launch_bounds__(256)
void cvt_transpose_bf16(const float* __restrict__ src, bf16_t* __restrict__ dst,
                        int rows, int cols) {
    __shared__ bf16_t tile[32][33];
    const int tx = threadIdx.x, ty = threadIdx.y;
    const int bx = blockIdx.x * 32, by = blockIdx.y * 32;
#pragma unroll
    for (int yy = ty; yy < 32; yy += 8)
        tile[yy][tx] = f2bf(src[(size_t)(by + yy) * cols + bx + tx]);
    __syncthreads();
#pragma unroll
    for (int yy = ty; yy < 32; yy += 8)
        dst[(size_t)(bx + yy) * rows + by + tx] = tile[tx][yy];
}

__global__ __launch_bounds__(256)
void rowsum_bf16(const bf16_t* __restrict__ g, float* __restrict__ sum, int rowlen) {
    __shared__ float red[256];
    const bf16_t* row = g + (size_t)blockIdx.x * rowlen;
    float s = 0.f;
    for (int c = threadIdx.x; c < rowlen; c += 256) s += bf2f(row[c]);
    red[threadIdx.x] = s;
    __syncthreads();
    for (int o = 128; o > 0; o >>= 1) {
        if ((int)threadIdx.x < o) red[threadIdx.x] += red[threadIdx.x + o];
        __syncthreads();
    }
    if (threadIdx.x == 0) sum[blockIdx.x] = red[0];
}

__global__ __launch_bounds__(256)
void gemv_m(const float* __restrict__ x, const float* __restrict__ w,
            const float* __restrict__ b, float* __restrict__ m, int cols) {
    __shared__ float red[256];
    int row = blockIdx.x;
    float s = 0.f;
    for (int c = threadIdx.x; c < cols; c += 256)
        s += x[(size_t)row * cols + c] * w[c];
    red[threadIdx.x] = s;
    __syncthreads();
    for (int o = 128; o > 0; o >>= 1) {
        if ((int)threadIdx.x < o) red[threadIdx.x] += red[threadIdx.x + o];
        __syncthreads();
    }
    if (threadIdx.x == 0) m[row] = red[0] + b[0];
}

__global__ __launch_bounds__(256)
void gate_softmax(const float* __restrict__ m, float* __restrict__ gate, int n) {
    __shared__ float red[256];
    __shared__ float s_max, s_sum;
    float mx = -3.4e38f;
    for (int i = threadIdx.x; i < n; i += 256) mx = fmaxf(mx, m[i]);
    red[threadIdx.x] = mx;
    __syncthreads();
    for (int o = 128; o > 0; o >>= 1) {
        if ((int)threadIdx.x < o) red[threadIdx.x] = fmaxf(red[threadIdx.x], red[threadIdx.x + o]);
        __syncthreads();
    }
    if (threadIdx.x == 0) s_max = red[0];
    __syncthreads();
    float sum = 0.f;
    for (int i = threadIdx.x; i < n; i += 256) sum += expf(m[i] - s_max);
    red[threadIdx.x] = sum;
    __syncthreads();
    for (int o = 128; o > 0; o >>= 1) {
        if ((int)threadIdx.x < o) red[threadIdx.x] += red[threadIdx.x + o];
        __syncthreads();
    }
    if (threadIdx.x == 0) s_sum = red[0];
    __syncthreads();
    for (int i = threadIdx.x; i < n; i += 256)
        gate[i] = expf(m[i] - s_max) / s_sum;
}

__global__ __launch_bounds__(256)
void row_stats_exp(bf16_t* __restrict__ S, float* __restrict__ zrow, int n) {
    __shared__ float red[256];
    __shared__ float s_max;
    bf16_t* srow = S + (size_t)blockIdx.x * n;
    float mx = -3.4e38f;
    for (int i = threadIdx.x; i < n; i += 256) mx = fmaxf(mx, bf2f(srow[i]));
    red[threadIdx.x] = mx;
    __syncthreads();
    for (int o = 128; o > 0; o >>= 1) {
        if ((int)threadIdx.x < o) red[threadIdx.x] = fmaxf(red[threadIdx.x], red[threadIdx.x + o]);
        __syncthreads();
    }
    if (threadIdx.x == 0) s_max = red[0];
    __syncthreads();
    float sum = 0.f;
    for (int i = threadIdx.x; i < n; i += 256) {
        bf16_t e = f2bf(expf(bf2f(srow[i]) - s_max));
        srow[i] = e;
        sum += bf2f(e);
    }
    red[threadIdx.x] = sum;
    __syncthreads();
    for (int o = 128; o > 0; o >>= 1) {
        if ((int)threadIdx.x < o) red[threadIdx.x] += red[threadIdx.x + o];
        __syncthreads();
    }
    if (threadIdx.x == 0) zrow[blockIdx.x] = red[0];
}

// ---------------------------------------------------------------------------
// Unified WMMA GEMM:  C[M x Ncols] = A[M x K] @ Bt[Ncols x K]^T
// B is ALWAYS [Ncols][K] row-major so staging is a contiguous b128 copy.
//
// Workgroup = 256 threads = 8 waves, 128x128 C tile; wave w owns
// row-tiles {2*(w/2), 2*(w/2)+1} x col-tiles {4*(w%2)..+3}.
// DOUBLE-BUFFERED LDS: next K-tile is DMA'd (global_load_async_to_lds_b128,
// ASYNCcnt) while the current tile feeds 8 WMMAs per wave.
//
// MODE: 0 = bf16 row-major store
//       1 = bf16 row-major store * (*scaleptr)
//       2 = f32 final epilogue: acc/zrow[row] + gate[row]*gsum[col] + x[row][col]
//       3 = bf16 TRANSPOSED store: Cb[col*ldT + row] (vectorized 16B stores)
// ---------------------------------------------------------------------------
template<int MODE>
__global__ __launch_bounds__(256)
void gemm_wmma(const bf16_t* __restrict__ A, const bf16_t* __restrict__ Bt,
               bf16_t* __restrict__ Cb, float* __restrict__ Cf,
               int Ncols, int K, int ldT,
               const float* __restrict__ scaleptr,
               const float* __restrict__ zrow,
               const float* __restrict__ gate,
               const float* __restrict__ gsum,
               const float* __restrict__ xres) {
    __shared__ bf16_t As[2][128][40];
    __shared__ bf16_t Bs[2][128][48];   // [colLocal][k]
    const int tid = threadIdx.x, wave = tid >> 5, lane = tid & 31;
    const int h = lane >> 4, ml = lane & 15;
    const int rt0 = (wave >> 1) * 2;
    const int ct0 = (wave & 1) * 4;
    const int row0 = blockIdx.y * 128, col0 = blockIdx.x * 128;

    const int ar = tid >> 1, as = (tid & 1) * 16;   // A staging: row, k-seg
    const int bj = tid >> 1, bs = (tid & 1) * 16;   // B staging: col, k-seg
    const bf16_t* aSrcBase = A  + (size_t)(row0 + ar) * K + as;
    const bf16_t* bSrcBase = Bt + (size_t)(col0 + bj) * K + bs;

    auto stage = [&](int buf, int kk) {
#if HAVE_ASYNC_LDS
        __builtin_amdgcn_global_load_async_to_lds_b128(
            as_glb(aSrcBase + kk),     as_lds(&As[buf][ar][as]),     0, 0);
        __builtin_amdgcn_global_load_async_to_lds_b128(
            as_glb(aSrcBase + kk + 8), as_lds(&As[buf][ar][as + 8]), 0, 0);
        __builtin_amdgcn_global_load_async_to_lds_b128(
            as_glb(bSrcBase + kk),     as_lds(&Bs[buf][bj][bs]),     0, 0);
        __builtin_amdgcn_global_load_async_to_lds_b128(
            as_glb(bSrcBase + kk + 8), as_lds(&Bs[buf][bj][bs + 8]), 0, 0);
#else
        *(v8bf*)&As[buf][ar][as]     = *(const v8bf*)(aSrcBase + kk);
        *(v8bf*)&As[buf][ar][as + 8] = *(const v8bf*)(aSrcBase + kk + 8);
        *(v8bf*)&Bs[buf][bj][bs]     = *(const v8bf*)(bSrcBase + kk);
        *(v8bf*)&Bs[buf][bj][bs + 8] = *(const v8bf*)(bSrcBase + kk + 8);
#endif
    };

    v8f acc[2][4] = {};
    stage(0, 0);
    for (int kk = 0; kk < K; kk += 32) {
        const int cur = (kk >> 5) & 1;
#if HAVE_ASYNC_LDS
        __builtin_amdgcn_s_wait_asynccnt(0);   // our writes to buf[cur] landed
#endif
        __syncthreads();                        // tile globally visible; prev reads of other buf done
        if (kk + 32 < K) stage(1 - cur, kk + 32);   // DMA next tile, overlaps WMMAs below

        v16bf a[2];
#pragma unroll
        for (int r = 0; r < 2; ++r)
#pragma unroll
            for (int v = 0; v < 8; ++v) {
                int k2 = (v < 4) ? (2 * v + 8 * h) : (16 + 2 * (v - 4) + 8 * h);
                a[r][2 * v]     = As[cur][(rt0 + r) * 16 + ml][k2];
                a[r][2 * v + 1] = As[cur][(rt0 + r) * 16 + ml][k2 + 1];
            }
#pragma unroll
        for (int c = 0; c < 4; ++c) {
            v16bf b;
#pragma unroll
            for (int e = 0; e < 16; ++e) b[e] = Bs[cur][(ct0 + c) * 16 + ml][(e < 8) ? (8 * h + e) : (16 + 8 * h + (e - 8))];
#pragma unroll
            for (int r = 0; r < 2; ++r)
                acc[r][c] = __builtin_amdgcn_wmma_f32_16x16x32_bf16(
                    false, a[r], false, b, (short)0, acc[r][c], false, false);
        }
        __syncthreads();   // all waves done reading buf[cur] before it is re-staged
    }
    // ---- epilogue ----
    float scale = (MODE == 1) ? *scaleptr : 1.f;
#pragma unroll
    for (int r = 0; r < 2; ++r)
#pragma unroll
        for (int c = 0; c < 4; ++c) {
            if (MODE == 3) {
                bf16_t tmp[8];
#pragma unroll
                for (int v = 0; v < 8; ++v) tmp[v] = f2bf(acc[r][c][v]);
                int col = col0 + (ct0 + c) * 16 + ml;
                int rowbase = row0 + (rt0 + r) * 16 + 8 * h;
                *(v8bf*)&Cb[(size_t)col * ldT + rowbase] = *(v8bf*)&tmp[0];
            } else {
#pragma unroll
                for (int v = 0; v < 8; ++v) {
                    int row = row0 + (rt0 + r) * 16 + v + 8 * h;
                    int col = col0 + (ct0 + c) * 16 + ml;
                    float val = acc[r][c][v];
                    if (MODE == 0) {
                        Cb[(size_t)row * Ncols + col] = f2bf(val);
                    } else if (MODE == 1) {
                        Cb[(size_t)row * Ncols + col] = f2bf(scale * val);
                    } else {
                        val = val / zrow[row] + gate[row] * gsum[col]
                            + xres[(size_t)row * Ncols + col];
                        Cf[(size_t)row * Ncols + col] = val;
                    }
                }
            }
        }
}

// ---------------------------------------------------------------------------
// Host-side orchestration
// ---------------------------------------------------------------------------
extern "C" void kernel_launch(void* const* d_in, const int* in_sizes, int n_in,
                              void* d_out, int out_size, void* d_ws, size_t ws_size,
                              hipStream_t stream) {
    const float* x    = (const float*)d_in[0];
    const float* Wq   = (const float*)d_in[1];
    const float* Wk   = (const float*)d_in[2];
    const float* Wg   = (const float*)d_in[3];
    const float* Wm_w = (const float*)d_in[4];
    const float* Wm_b = (const float*)d_in[5];
    const float* cptr = (const float*)d_in[6];
    float* out = (float*)d_out;

    const int N = N_TOK, D = DIM;

    char* p = (char*)d_ws;
    auto carve = [&](size_t bytes) -> char* {
        char* r = p;
        p += (bytes + 255) & ~(size_t)255;
        return r;
    };
    float*  xmean = (float*)carve(sizeof(float) * D);
    float*  gsum  = (float*)carve(sizeof(float) * D);
    float*  mvec  = (float*)carve(sizeof(float) * N);
    float*  gate  = (float*)carve(sizeof(float) * N);
    float*  zrow  = (float*)carve(sizeof(float) * N);
    bf16_t* xb    = (bf16_t*)carve(sizeof(bf16_t) * (size_t)N * D);
    bf16_t* xcb   = (bf16_t*)carve(sizeof(bf16_t) * (size_t)N * D);
    bf16_t* wqT   = (bf16_t*)carve(sizeof(bf16_t) * (size_t)D * D);
    bf16_t* wkT   = (bf16_t*)carve(sizeof(bf16_t) * (size_t)D * D);
    bf16_t* wgT   = (bf16_t*)carve(sizeof(bf16_t) * (size_t)D * D);
    bf16_t* qcb   = (bf16_t*)carve(sizeof(bf16_t) * (size_t)N * D);
    bf16_t* kcb   = (bf16_t*)carve(sizeof(bf16_t) * (size_t)N * D);
    bf16_t* gT    = (bf16_t*)carve(sizeof(bf16_t) * (size_t)N * D);  // [D][N]
    bf16_t* S     = (bf16_t*)carve(sizeof(bf16_t) * (size_t)N * N);

    // 1) column mean of x; centered/raw bf16 copies of x
    colmean_f32<<<dim3((D + 255) / 256), 256, 0, stream>>>(x, xmean, N, D);
    prep_x<<<dim3(2048), 256, 0, stream>>>(x, xmean, xb, xcb, N, D);

    // 2) weights -> bf16, transposed ([outCol][k]) so GEMM B staging is contiguous
    dim3 tgrid(D / 32, D / 32), tblk(32, 8);
    cvt_transpose_bf16<<<tgrid, tblk, 0, stream>>>(Wq, wqT, D, D);
    cvt_transpose_bf16<<<tgrid, tblk, 0, stream>>>(Wk, wkT, D, D);
    cvt_transpose_bf16<<<tgrid, tblk, 0, stream>>>(Wg, wgT, D, D);

    // 3) projections: qc = xc@Wq, kc = xc@Wk (centering commutes with proj),
    //    gT = (x@Wg)^T written transposed for the final GEMM's B path
    dim3 gproj(D / 128, N / 128);
    gemm_wmma<0><<<gproj, 256, 0, stream>>>(xcb, wqT, qcb, nullptr, D, D, 0,
                                            nullptr, nullptr, nullptr, nullptr, nullptr);
    gemm_wmma<0><<<gproj, 256, 0, stream>>>(xcb, wkT, kcb, nullptr, D, D, 0,
                                            nullptr, nullptr, nullptr, nullptr, nullptr);
    gemm_wmma<3><<<gproj, 256, 0, stream>>>(xb, wgT, gT, nullptr, D, D, N,
                                            nullptr, nullptr, nullptr, nullptr, nullptr);

    // 4) gsum[d] = rowsum of gT; gate softmax over tokens
    rowsum_bf16<<<dim3(D), 256, 0, stream>>>(gT, gsum, N);
    gemv_m<<<dim3(N), 256, 0, stream>>>(x, Wm_w, Wm_b, mvec, D);
    gate_softmax<<<dim3(1), 256, 0, stream>>>(mvec, gate, N);

    // 5) scores S = c * qc @ kc^T
    dim3 gs(N / 128, N / 128);
    gemm_wmma<1><<<gs, 256, 0, stream>>>(qcb, kcb, S, nullptr, N, D, 0,
                                         cptr, nullptr, nullptr, nullptr, nullptr);

    // 6) per-row softmax stats; S <- exp(S - rowmax), Z per row
    row_stats_exp<<<dim3(N), 256, 0, stream>>>(S, zrow, N);

    // 7) out = (U @ g)/Z + gate*gsum + x   (B = gT, K = N)
    dim3 gfin(D / 128, N / 128);
    gemm_wmma<2><<<gfin, 256, 0, stream>>>(S, gT, nullptr, out, D, N, 0,
                                           nullptr, zrow, gate, gsum, x);
}
